// MultyHeadAttn_20486994002756
// MI455X (gfx1250) — matrix operand order repaired and checked by reading, not compiled, not run on hardware
//
#include <hip/hip_runtime.h>
#include <math.h>

// ---------------------------------------------------------------------------
// MultyHeadAttn for MI455X (gfx1250, wave32, WMMA + async-to-LDS).
// Pipeline: 3x (GEMM1+bias -> LN -> GELU -> GEMM2+bias) -> QK^T*scale ->
//           masked softmax -> attn@V -> LN + residual.
// All GEMMs: bf16 operands (pre-converted / pre-transposed so both A and B
// are row-major N-x-K "direct" layout), fp32 accumulation via
// v_wmma_f32_16x16x32_bf16. Tiles are staged into double-buffered LDS with
// global_load_async_to_lds_b128 (ASYNCcnt pipeline, 2 tiles deep).
// ---------------------------------------------------------------------------

typedef __bf16 bf16_t;
typedef bf16_t v16bf __attribute__((ext_vector_type(16)));
typedef float  v8f   __attribute__((ext_vector_type(8)));

__device__ __forceinline__ bf16_t f2bf(float f) {
    unsigned u = __float_as_uint(f);
    u += 0x7FFFu + ((u >> 16) & 1u);               // round-to-nearest-even
    unsigned short s = (unsigned short)(u >> 16);
    return __builtin_bit_cast(bf16_t, s);
}
__device__ __forceinline__ float bf2f(bf16_t b) {
    unsigned short s = __builtin_bit_cast(unsigned short, b);
    return __uint_as_float(((unsigned)s) << 16);
}
__device__ __forceinline__ float to_f(float x)  { return x; }
__device__ __forceinline__ float to_f(bf16_t x) { return bf2f(x); }
__device__ __forceinline__ void  st_c(float* p, float v)  { *p = v; }
__device__ __forceinline__ void  st_c(bf16_t* p, float v) { *p = f2bf(v); }

union H8   { bf16_t h[8];  uint4 u;    };          // 16 bytes
union FragU{ v16bf  v;     uint4 q[2]; };          // 32 bytes (8 VGPRs)

#define LDS_STRIDE 40   // 32 bf16 + 8 pad (80B rows -> spread LDS banks)

// ---------------------------------------------------------------------------
// bf16 WMMA GEMM:  C[z] = scale * A[z] x B[z]^T + bias      (B is N x K)
//   Block tile 128x128, K-step 32; 256 threads = 8 wave32s (4m x 2n grid),
//   each wave owns 32x64 = 2x4 v_wmma accumulators.
//   Staging: global_load_async_to_lds_b128, double-buffered, 2-deep pipeline.
//   Requires M,N % 128 == 0, K % 32 == 0 (true for every call here).
// ---------------------------------------------------------------------------
template <typename CT>
__global__ __launch_bounds__(256)
void gemm_bf16_wmma_kernel(CT* __restrict__ C, const bf16_t* __restrict__ A,
                           const bf16_t* __restrict__ Bm,
                           const float* __restrict__ bias,
                           int M, int N, int K, int lda, int ldb, int ldc,
                           float scale, size_t batchA, size_t batchB, size_t batchC)
{
    __shared__ __attribute__((aligned(16))) bf16_t lA[2][128 * LDS_STRIDE];
    __shared__ __attribute__((aligned(16))) bf16_t lB[2][128 * LDS_STRIDE];

    const int tid  = threadIdx.x;
    const int lane = tid & 31;
    const int wave = tid >> 5;
    const int wrow = wave & 3;          // 4 wave-rows * 32 = 128 M
    const int wcol = wave >> 2;         // 2 wave-cols * 64 = 128 N
    const int hlf  = lane >> 4;         // lane half (0/1)
    const int l16  = lane & 15;

    const int z = blockIdx.z;
    A  += (size_t)z * batchA;
    Bm += (size_t)z * batchB;
    C  += (size_t)z * batchC;
    const int m0 = blockIdx.y * 128;
    const int n0 = blockIdx.x * 128;

    // staging coordinates: each thread copies 2x16B of A and 2x16B of B;
    // per wave that is 4 async b128 issues (ASYNCcnt += 4 per tile).
    const int rr = tid >> 2;            // 0..63
    const int cc = (tid & 3) * 8;       // 0,8,16,24 (bf16 units)

    auto issue_tile = [&](int buf, int k0) {
        {
            unsigned       l0 = (unsigned)(size_t)&lA[buf][rr * LDS_STRIDE + cc];
            const bf16_t*  g0 = A + (size_t)(m0 + rr) * lda + (k0 + cc);
            asm volatile("global_load_async_to_lds_b128 %0, %1, off"
                         :: "v"(l0), "v"(g0) : "memory");
            unsigned       l1 = (unsigned)(size_t)&lA[buf][(rr + 64) * LDS_STRIDE + cc];
            const bf16_t*  g1 = A + (size_t)(m0 + rr + 64) * lda + (k0 + cc);
            asm volatile("global_load_async_to_lds_b128 %0, %1, off"
                         :: "v"(l1), "v"(g1) : "memory");
        }
        {
            unsigned       l0 = (unsigned)(size_t)&lB[buf][rr * LDS_STRIDE + cc];
            const bf16_t*  g0 = Bm + (size_t)(n0 + rr) * ldb + (k0 + cc);
            asm volatile("global_load_async_to_lds_b128 %0, %1, off"
                         :: "v"(l0), "v"(g0) : "memory");
            unsigned       l1 = (unsigned)(size_t)&lB[buf][(rr + 64) * LDS_STRIDE + cc];
            const bf16_t*  g1 = Bm + (size_t)(n0 + rr + 64) * ldb + (k0 + cc);
            asm volatile("global_load_async_to_lds_b128 %0, %1, off"
                         :: "v"(l1), "v"(g1) : "memory");
        }
    };

    v8f acc[2][4];
    for (int i = 0; i < 2; ++i)
        for (int j = 0; j < 4; ++j)
            for (int r = 0; r < 8; ++r) acc[i][j][r] = 0.0f;

    issue_tile(0, 0);
    if (K > 32) issue_tile(1, 32);

    for (int k0 = 0, s = 0; k0 < K; k0 += 32, ++s) {
        // retire the oldest async group (in-order per wave): <=4 keeps the
        // prefetched group in flight; drain fully on the last step.
        if (k0 + 32 < K) asm volatile("s_wait_asynccnt 0x4" ::: "memory");
        else             asm volatile("s_wait_asynccnt 0x0" ::: "memory");
        __syncthreads();                 // tile s visible to every wave

        const bf16_t* bA = lA[s & 1];
        const bf16_t* bB = lB[s & 1];

        // A 16x32 bf16 frag (ISA 7.12.2): lanes0-15 k=0..7 & 16..23,
        //                                 lanes16-31 k=8..15 & 24..31.
        FragU a[2], b[4];
        #pragma unroll
        for (int i = 0; i < 2; ++i) {
            const int m = wrow * 32 + i * 16 + l16;
            a[i].q[0] = *(const uint4*)&bA[m * LDS_STRIDE + hlf * 8];
            a[i].q[1] = *(const uint4*)&bA[m * LDS_STRIDE + hlf * 8 + 16];
        }
        // B 32x16 bf16 frag: lanes0-15 k=0..7 & 16..23, lanes16-31 k=8..15 & 24..31.
        #pragma unroll
        for (int j = 0; j < 4; ++j) {
            const int n = wcol * 64 + j * 16 + l16;
            b[j].q[0] = *(const uint4*)&bB[n * LDS_STRIDE + hlf * 8];
            b[j].q[1] = *(const uint4*)&bB[n * LDS_STRIDE + hlf * 8 + 16];
        }
        #pragma unroll
        for (int i = 0; i < 2; ++i)
            #pragma unroll
            for (int j = 0; j < 4; ++j)
                acc[i][j] = __builtin_amdgcn_wmma_f32_16x16x32_bf16(
                    false, a[i].v, false, b[j].v, (short)0, acc[i][j],
                    false, false);

        __syncthreads();                 // all reads of buffer s done
        if (k0 + 64 < K) issue_tile(s & 1, k0 + 64);   // prefetch tile s+2
    }

    // epilogue: C/D layout: VGPR r -> M = r + 8*half, N = lane&15
    #pragma unroll
    for (int i = 0; i < 2; ++i) {
        #pragma unroll
        for (int j = 0; j < 4; ++j) {
            const int n  = n0 + wcol * 64 + j * 16 + l16;
            const float bv = bias ? bias[n] : 0.0f;
            #pragma unroll
            for (int r = 0; r < 8; ++r) {
                const int m = m0 + wrow * 32 + i * 16 + hlf * 8 + r;
                st_c(&C[(size_t)m * ldc + n], acc[i][j][r] * scale + bv);
            }
        }
    }
}

// ---------------------------------------------------------------------------
// fp32 -> bf16 elementwise convert (8 elems/thread, 16B stores).
// ---------------------------------------------------------------------------
__global__ __launch_bounds__(256)
void cvt_bf16_kernel(bf16_t* __restrict__ out, const float* __restrict__ in,
                     size_t n)
{
    const size_t base = ((size_t)blockIdx.x * 256 + threadIdx.x) * 8;
    if (base + 8 > n) return;
    H8 vv;
    #pragma unroll
    for (int t = 0; t < 8; ++t) vv.h[t] = f2bf(in[base + t]);
    *(uint4*)&out[base] = vv.u;
}

// ---------------------------------------------------------------------------
// Tiled transpose (+convert) to bf16: in R x C  ->  out C x R, per batch z.
// ---------------------------------------------------------------------------
template <typename IT>
__global__ __launch_bounds__(256)
void transpose_to_bf16_kernel(bf16_t* __restrict__ out, const IT* __restrict__ in,
                              int R, int Ccols)
{
    __shared__ bf16_t t[32][33];
    const int z = blockIdx.z;
    in  += (size_t)z * R * Ccols;
    out += (size_t)z * R * Ccols;
    const int c0 = blockIdx.x * 32;
    const int r0 = blockIdx.y * 32;
    const int lx = threadIdx.x & 31;
    const int ly = threadIdx.x >> 5;      // 0..7
    for (int i = ly; i < 32; i += 8)
        t[i][lx] = f2bf(to_f(in[(size_t)(r0 + i) * Ccols + (c0 + lx)]));
    __syncthreads();
    for (int i = ly; i < 32; i += 8)
        out[(size_t)(c0 + i) * R + (r0 + lx)] = t[lx][i];
}

// ---------------------------------------------------------------------------
// Row-wise LayerNorm + exact GELU, fp32 in -> bf16 out. One block per row.
// ---------------------------------------------------------------------------
__global__ __launch_bounds__(256)
void ln_gelu_kernel(bf16_t* __restrict__ out, const float* __restrict__ in,
                    const float* __restrict__ g, const float* __restrict__ be,
                    int D)
{
    __shared__ float sred[256];
    const size_t row = blockIdx.x;
    const float* x = in + row * (size_t)D;
    bf16_t* o      = out + row * (size_t)D;
    const int tid = threadIdx.x;

    float s1 = 0.f, s2 = 0.f;
    for (int d = tid; d < D; d += 256) { float v = x[d]; s1 += v; s2 += v * v; }
    sred[tid] = s1; __syncthreads();
    for (int s = 128; s; s >>= 1) { if (tid < s) sred[tid] += sred[tid + s]; __syncthreads(); }
    const float mean = sred[0] / (float)D; __syncthreads();
    sred[tid] = s2; __syncthreads();
    for (int s = 128; s; s >>= 1) { if (tid < s) sred[tid] += sred[tid + s]; __syncthreads(); }
    const float var  = sred[0] / (float)D - mean * mean;
    const float rstd = rsqrtf(var + 1e-5f);

    for (int d = tid; d < D; d += 256) {
        float v = (x[d] - mean) * rstd * g[d] + be[d];
        v = 0.5f * v * (1.0f + erff(v * 0.70710678118654752f));
        o[d] = f2bf(v);
    }
}

// ---------------------------------------------------------------------------
// Masked softmax over K axis, fp32 scores -> bf16 attn. Faithful reference:
//   s = mask ? x : -inf ; softmax(s) ; *= (1 - mask)   (NaN rows if mask==0)
// One block per (b,q) row; Kd <= 2048 held in registers (8 per lane).
// ---------------------------------------------------------------------------
__global__ __launch_bounds__(256)
void softmax_mask_kernel(bf16_t* __restrict__ attn, const float* __restrict__ sc,
                         const int* __restrict__ mask, int Kd)
{
    __shared__ float sred[256];
    const size_t row = blockIdx.x;
    const float* x = sc   + row * (size_t)Kd;
    bf16_t* o      = attn + row * (size_t)Kd;
    const int tid = threadIdx.x;
    const int mv  = mask[row];
    const float keep = mv ? 0.0f : -__builtin_inff();
    const float post = mv ? 0.0f : 1.0f;

    float v[8];
    float mx = -__builtin_inff();
    #pragma unroll
    for (int t = 0; t < 8; ++t) {
        const int d = tid + t * 256;
        v[t] = (d < Kd) ? (x[d] + keep) : -__builtin_inff();
        mx = fmaxf(mx, v[t]);
    }
    sred[tid] = mx; __syncthreads();
    for (int s = 128; s; s >>= 1) { if (tid < s) sred[tid] = fmaxf(sred[tid], sred[tid + s]); __syncthreads(); }
    mx = sred[0]; __syncthreads();

    float sum = 0.f;
    #pragma unroll
    for (int t = 0; t < 8; ++t) { v[t] = expf(v[t] - mx); sum += v[t]; }
    sred[tid] = sum; __syncthreads();
    for (int s = 128; s; s >>= 1) { if (tid < s) sred[tid] += sred[tid + s]; __syncthreads(); }
    const float inv = post / sred[0];

    #pragma unroll
    for (int t = 0; t < 8; ++t) {
        const int d = tid + t * 256;
        if (d < Kd) o[d] = f2bf(v[t] * inv);
    }
}

// ---------------------------------------------------------------------------
// Final LayerNorm + residual: out = resid + LN(x)*g + be. One block per row.
// ---------------------------------------------------------------------------
__global__ __launch_bounds__(256)
void ln_residual_kernel(float* __restrict__ out, const float* __restrict__ in,
                        const float* __restrict__ resid,
                        const float* __restrict__ g, const float* __restrict__ be,
                        int D)
{
    __shared__ float sred[256];
    const size_t row = blockIdx.x;
    const float* x = in + row * (size_t)D;
    const float* r = resid + row * (size_t)D;
    float* o       = out + row * (size_t)D;
    const int tid = threadIdx.x;

    float s1 = 0.f, s2 = 0.f;
    for (int d = tid; d < D; d += 256) { float v = x[d]; s1 += v; s2 += v * v; }
    sred[tid] = s1; __syncthreads();
    for (int s = 128; s; s >>= 1) { if (tid < s) sred[tid] += sred[tid + s]; __syncthreads(); }
    const float mean = sred[0] / (float)D; __syncthreads();
    sred[tid] = s2; __syncthreads();
    for (int s = 128; s; s >>= 1) { if (tid < s) sred[tid] += sred[tid + s]; __syncthreads(); }
    const float var  = sred[0] / (float)D - mean * mean;
    const float rstd = rsqrtf(var + 1e-5f);

    for (int d = tid; d < D; d += 256)
        o[d] = r[d] + (x[d] - mean) * rstd * g[d] + be[d];
}

// ---------------------------------------------------------------------------
extern "C" void kernel_launch(void* const* d_in, const int* in_sizes, int n_in,
                              void* d_out, int out_size, void* d_ws, size_t ws_size,
                              hipStream_t stream)
{
    (void)in_sizes; (void)n_in; (void)out_size; (void)ws_size;

    constexpr int B = 16, Q = 2048, KS = 2048, D = 1024;
    constexpr size_t NT = (size_t)B * Q * D;    // 33,554,432 elems
    constexpr size_t NS = (size_t)B * Q * KS;   // 67,108,864 elems
    constexpr size_t ND = (size_t)D * D;        // 1,048,576 elems

    const float* qin  = (const float*)d_in[0];
    const float* kin  = (const float*)d_in[1];
    const float* vin  = (const float*)d_in[2];
    const int*   mask = (const int*)d_in[3];
    const float* W1[3] = {(const float*)d_in[4],  (const float*)d_in[10], (const float*)d_in[16]};
    const float* b1[3] = {(const float*)d_in[5],  (const float*)d_in[11], (const float*)d_in[17]};
    const float* gg[3] = {(const float*)d_in[6],  (const float*)d_in[12], (const float*)d_in[18]};
    const float* bb[3] = {(const float*)d_in[7],  (const float*)d_in[13], (const float*)d_in[19]};
    const float* W2[3] = {(const float*)d_in[8],  (const float*)d_in[14], (const float*)d_in[20]};
    const float* b2[3] = {(const float*)d_in[9],  (const float*)d_in[15], (const float*)d_in[21]};
    const float* out_g  = (const float*)d_in[22];
    const float* out_be = (const float*)d_in[23];
    const float* X[3] = {qin, kin, vin};

    // ---- workspace carve (~1.04 GiB) --------------------------------------
    char* ws = (char*)d_ws;
    size_t off = 0;
    auto carve = [&](size_t bytes) -> char* {
        char* p = ws + off; off += (bytes + 255) & ~(size_t)255; return p;
    };
    bf16_t* Wt1[3]; bf16_t* Wt2[3];
    for (int p = 0; p < 3; ++p) { Wt1[p] = (bf16_t*)carve(ND * 2);
                                  Wt2[p] = (bf16_t*)carve(ND * 2); }
    bf16_t* P[3];
    for (int p = 0; p < 3; ++p) P[p] = (bf16_t*)carve(NT * 2);
    bf16_t* xbf  = (bf16_t*)carve(NT * 2);   // bf16 copy of current input
    float*  h    = (float*) carve(NT * 4);   // GEMM1 output (fp32 for LN)
    bf16_t* hg   = (bf16_t*)carve(NT * 2);   // LN+GELU output
    bf16_t* VpT  = (bf16_t*)carve(NT * 2);   // V projection, [b][d][k]
    float*  sc   = (float*) carve(NS * 4);   // fp32 scores
    bf16_t* attn = (bf16_t*)carve(NS * 2);   // bf16 attention weights
    float*  xo   = (float*) carve(NT * 4);   // attn @ V (fp32 for LN)

    const dim3 blk(256);
    const int  Mp = B * Q;                   // 32768 rows

    // ---- pre-pass: weights -> bf16, transposed to N x K -------------------
    for (int p = 0; p < 3; ++p) {
        transpose_to_bf16_kernel<float><<<dim3(D / 32, D / 32, 1), blk, 0, stream>>>(
            Wt1[p], W1[p], D, D);
        transpose_to_bf16_kernel<float><<<dim3(D / 32, D / 32, 1), blk, 0, stream>>>(
            Wt2[p], W2[p], D, D);
    }

    // ---- projections: x@W1+b1 -> LN -> GELU -> @W2+b2 (bf16 out) ----------
    for (int p = 0; p < 3; ++p) {
        cvt_bf16_kernel<<<dim3(NT / (256 * 8)), blk, 0, stream>>>(xbf, X[p], NT);
        gemm_bf16_wmma_kernel<float>
            <<<dim3(D / 128, Mp / 128, 1), blk, 0, stream>>>(
                h, xbf, Wt1[p], b1[p], Mp, D, D, D, D, D, 1.0f, 0, 0, 0);
        ln_gelu_kernel<<<dim3(Mp), blk, 0, stream>>>(hg, h, gg[p], bb[p], D);
        gemm_bf16_wmma_kernel<bf16_t>
            <<<dim3(D / 128, Mp / 128, 1), blk, 0, stream>>>(
                P[p], hg, Wt2[p], b2[p], Mp, D, D, D, D, D, 1.0f, 0, 0, 0);
    }

    // ---- V projection -> [b][d][k] for direct N x K staging ---------------
    transpose_to_bf16_kernel<bf16_t><<<dim3(D / 32, KS / 32, B), blk, 0, stream>>>(
        VpT, P[2], KS, D);

    // ---- scores = Qp @ Kp^T * (1/sqrt(D)) ---------------------------------
    gemm_bf16_wmma_kernel<float>
        <<<dim3(KS / 128, Q / 128, B), blk, 0, stream>>>(
            sc, P[0], P[1], nullptr, Q, KS, D, D, D, KS, 0.03125f,
            (size_t)Q * D, (size_t)KS * D, (size_t)Q * KS);

    softmax_mask_kernel<<<dim3(B * Q), blk, 0, stream>>>(attn, sc, mask, KS);

    // ---- x = attn @ Vp  (B operand = VpT, already N x K) ------------------
    gemm_bf16_wmma_kernel<float>
        <<<dim3(D / 128, Q / 128, B), blk, 0, stream>>>(
            xo, attn, VpT, nullptr, Q, D, KS, KS, KS, D, 1.0f,
            (size_t)Q * KS, (size_t)D * KS, (size_t)Q * D);

    // ---- out = q + LN(x)*out_g + out_be -----------------------------------
    ln_residual_kernel<<<dim3(Mp), blk, 0, stream>>>(
        (float*)d_out, xo, qin, out_g, out_be, D);
}
